// MultiHeadAttention_34205119545583
// MI455X (gfx1250) — matrix operand order repaired and checked by reading, not compiled, not run on hardware
//
#include <hip/hip_runtime.h>

#define B_    64
#define S_    512
#define H_    128
#define NH_   4
#define NHH_  512       // NH*H
#define BS_   32768     // B*S
#define LN_EPS 1e-6f
#define NEG_INF_ -1e9f

typedef __attribute__((ext_vector_type(16))) __bf16 v16bf;
typedef __attribute__((ext_vector_type(8)))  __bf16 v8bf;
typedef __attribute__((ext_vector_type(8)))  float  v8f;

__device__ __forceinline__ v16bf frag_concat(v8bf lo, v8bf hi) {
  return __builtin_shufflevector(lo, hi, 0,1,2,3,4,5,6,7,8,9,10,11,12,13,14,15);
}

// A-matrix (16xK=32) bf16 fragment from row-major storage.
// lane = m + 16*hi ; element e -> row m, K = k0 + (e&7) + 8*hi + 16*(e>>3)
__device__ __forceinline__ v16bf load_a(const __bf16* base, int ld, int m, int hi, int k0) {
  const v8bf* p = (const v8bf*)(base + (size_t)m * ld + k0 + 8 * hi);
  return frag_concat(p[0], p[2]);
}

// B-matrix (K=32 x 16) bf16 fragment from *transposed* (N-major, K-contiguous) storage.
// lane = n + 16*hi ; element e -> col n, K = k0 + (e&7) + 8*hi + 16*(e>>3)
__device__ __forceinline__ v16bf load_bT(const __bf16* baseT, int ld, int n, int hi, int k0) {
  const v8bf* p = (const v8bf*)(baseT + (size_t)n * ld + k0 + 8 * hi);
  return frag_concat(p[0], p[2]);
}

__device__ __forceinline__ v8f wmma_bf16(v16bf a, v16bf b, v8f c) {
  return __builtin_amdgcn_wmma_f32_16x16x32_bf16(false, a, false, b, (short)0, c, false, false);
}

// ---------------- pack / convert kernels ----------------
__global__ void k_cast(const float* __restrict__ src, __bf16* __restrict__ dst, int n) {
  int i = blockIdx.x * blockDim.x + threadIdx.x;
  if (i < n) dst[i] = (__bf16)src[i];
}

// dst[c*rows + r] = (bf16)src[r*cols + c]
__global__ void k_tcast(const float* __restrict__ src, __bf16* __restrict__ dst, int rows, int cols) {
  int i = blockIdx.x * blockDim.x + threadIdx.x;
  if (i < rows * cols) {
    int r = i / cols, c = i - r * cols;
    dst[(size_t)c * rows + r] = (__bf16)src[i];
  }
}

// ---------------- QKV projection ----------------
// grid: (BS/64, NHH/64, 3); block: 128 (4 waves, each 16 rows x 64 cols)
__global__ void __launch_bounds__(128) k_qkv(const __bf16* __restrict__ encB,
                                             const __bf16* __restrict__ WqT,
                                             const __bf16* __restrict__ WkT,
                                             const __bf16* __restrict__ WvT,
                                             __bf16* __restrict__ Qb,
                                             __bf16* __restrict__ Kb,
                                             __bf16* __restrict__ Vt) {
  const int lane = threadIdx.x & 31;
  const int wave = threadIdx.x >> 5;
  const int lm = lane & 15, hi = lane >> 4;
  const int m0 = blockIdx.x * 64 + wave * 16;
  const int n0 = blockIdx.y * 64;
  const int proj = blockIdx.z;
  const __bf16* WT = (proj == 0) ? WqT : (proj == 1 ? WkT : WvT);
  const __bf16* Abase = encB + (size_t)m0 * H_;

  v8f acc[4] = {};
  for (int k0 = 0; k0 < H_; k0 += 32) {
    v16bf a = load_a(Abase, H_, lm, hi, k0);
#pragma unroll
    for (int j = 0; j < 4; ++j) {
      v16bf b = load_bT(WT + (size_t)(n0 + j * 16) * H_, H_, lm, hi, k0);
      acc[j] = wmma_bf16(a, b, acc[j]);
    }
  }
  const int bIdx = m0 >> 9;      // / S_
  const int sBase = m0 & 511;    // % S_
#pragma unroll
  for (int j = 0; j < 4; ++j) {
    int n = n0 + j * 16 + lm;
    int h = n >> 7, d = n & 127;
#pragma unroll
    for (int r = 0; r < 8; ++r) {
      int s = sBase + r + 8 * hi;
      __bf16 val = (__bf16)acc[j][r];
      size_t bh = (size_t)bIdx * NH_ + h;
      if (proj == 0)      Qb[(bh * S_ + s) * H_ + d] = val;
      else if (proj == 1) Kb[(bh * S_ + s) * H_ + d] = val;
      else                Vt[(bh * H_ + d) * S_ + s] = val;   // transposed
    }
  }
}

// ---------------- flash attention ----------------
// grid: (S/64, NH, B); block 128 (4 waves, each owns 16 query rows)
__global__ void __launch_bounds__(128) k_attn(const __bf16* __restrict__ Qb,
                                              const __bf16* __restrict__ Kb,
                                              const __bf16* __restrict__ Vt,
                                              const int* __restrict__ mask,
                                              __bf16* __restrict__ Ctxb) {
  __shared__ alignas(16) __bf16 Plds[4][16][32];
  const int lane = threadIdx.x & 31;
  const int wave = threadIdx.x >> 5;
  const int lm = lane & 15, hi = lane >> 4;
  const int q0 = blockIdx.x * 64 + wave * 16;
  const int h = blockIdx.y;
  const int b = blockIdx.z;
  const size_t bh = (size_t)b * NH_ + h;
  const __bf16* Q = Qb + (bh * S_ + q0) * H_;
  const __bf16* K = Kb + bh * S_ * H_;
  const __bf16* V = Vt + bh * (size_t)H_ * S_;
  const int* Mrow = mask + ((size_t)b * S_ + q0) * S_;
  const float scale = 0.08838834764831845f;  // 1/sqrt(128)

  v16bf qf[4];
#pragma unroll
  for (int t = 0; t < 4; ++t) qf[t] = load_a(Q, H_, lm, hi, t * 32);

  v8f O[8] = {};
  float mrow[8], lrow[8], corr[8];
#pragma unroll
  for (int r = 0; r < 8; ++r) { mrow[r] = -3e38f; lrow[r] = 0.f; }

  for (int kc = 0; kc < S_; kc += 32) {
    // scores for 32 keys: two 16-wide tiles, K-dim = 128
    v8f s0 = {}, s1 = {};
#pragma unroll
    for (int t = 0; t < 4; ++t) {
      s0 = wmma_bf16(qf[t], load_bT(K + (size_t)kc * H_,        H_, lm, hi, t * 32), s0);
      s1 = wmma_bf16(qf[t], load_bT(K + (size_t)(kc + 16) * H_, H_, lm, hi, t * 32), s1);
    }
    // mask + scale + online softmax (rows r+8hi live in 16-lane groups)
#pragma unroll
    for (int r = 0; r < 8; ++r) {
      const int* mr = Mrow + (size_t)(r + 8 * hi) * S_ + kc;
      s0[r] = mr[lm]      ? s0[r] * scale : NEG_INF_;
      s1[r] = mr[lm + 16] ? s1[r] * scale : NEG_INF_;
      float tm = fmaxf(s0[r], s1[r]);
#pragma unroll
      for (int d = 1; d < 16; d <<= 1) tm = fmaxf(tm, __shfl_xor(tm, d, 32));
      float nm = fmaxf(mrow[r], tm);
      corr[r] = __expf(mrow[r] - nm);
      mrow[r] = nm;
      float p0 = __expf(s0[r] - nm);
      float p1 = __expf(s1[r] - nm);
      s0[r] = p0; s1[r] = p1;
      float ps = p0 + p1;
#pragma unroll
      for (int d = 1; d < 16; d <<= 1) ps += __shfl_xor(ps, d, 32);
      lrow[r] = lrow[r] * corr[r] + ps;
    }
    // C-layout -> A-layout via LDS round trip (bf16 P tile, 16x32)
    __syncthreads();   // previous iteration's P reads complete before overwrite
#pragma unroll
    for (int r = 0; r < 8; ++r) {
      Plds[wave][r + 8 * hi][lm]      = (__bf16)s0[r];
      Plds[wave][r + 8 * hi][lm + 16] = (__bf16)s1[r];
    }
    __syncthreads();   // stores visible before fragment reload
    v16bf pa = load_a(&Plds[wave][0][0], 32, lm, hi, 0);
    // O = O*corr + P @ V   (V transposed: dim-major, key-contiguous)
#pragma unroll
    for (int j = 0; j < 8; ++j) {
#pragma unroll
      for (int r = 0; r < 8; ++r) O[j][r] *= corr[r];
      v16bf vb = load_bT(V + (size_t)(j * 16) * S_, S_, lm, hi, kc);
      O[j] = wmma_bf16(pa, vb, O[j]);
    }
  }
  // normalize and emit ctx as (B, S, NH*H) bf16
#pragma unroll
  for (int j = 0; j < 8; ++j) {
    int d = j * 16 + lm;
#pragma unroll
    for (int r = 0; r < 8; ++r) {
      int q = q0 + r + 8 * hi;
      Ctxb[((size_t)b * S_ + q) * NHH_ + h * H_ + d] = (__bf16)(O[j][r] / lrow[r]);
    }
  }
}

// ---------------- out-projection + residual + LayerNorm ----------------
// grid: BS/64; block 128 (4 waves, each 16 rows x 128 cols)
__global__ void __launch_bounds__(128) k_out(const __bf16* __restrict__ Ctxb,
                                             const __bf16* __restrict__ WoT,
                                             const float* __restrict__ enc,
                                             const float* __restrict__ gamma,
                                             const float* __restrict__ beta,
                                             float* __restrict__ out) {
  const int lane = threadIdx.x & 31;
  const int wave = threadIdx.x >> 5;
  const int lm = lane & 15, hi = lane >> 4;
  const int m0 = blockIdx.x * 64 + wave * 16;
  const __bf16* Abase = Ctxb + (size_t)m0 * NHH_;

  v8f acc[8] = {};
  for (int k0 = 0; k0 < NHH_; k0 += 32) {
    v16bf a = load_a(Abase, NHH_, lm, hi, k0);
#pragma unroll
    for (int j = 0; j < 8; ++j) {
      v16bf b = load_bT(WoT + (size_t)(j * 16) * NHH_, NHH_, lm, hi, k0);
      acc[j] = wmma_bf16(a, b, acc[j]);
    }
  }
  // residual
#pragma unroll
  for (int j = 0; j < 8; ++j) {
    int n = j * 16 + lm;
#pragma unroll
    for (int r = 0; r < 8; ++r)
      acc[j][r] += enc[(size_t)(m0 + r + 8 * hi) * H_ + n];
  }
  // LayerNorm over 128 cols: each row's values live in one 16-lane group, 8 per lane
#pragma unroll
  for (int r = 0; r < 8; ++r) {
    float s = 0.f, sq = 0.f;
#pragma unroll
    for (int j = 0; j < 8; ++j) { float v = acc[j][r]; s += v; sq += v * v; }
#pragma unroll
    for (int d = 1; d < 16; d <<= 1) { s += __shfl_xor(s, d, 32); sq += __shfl_xor(sq, d, 32); }
    float mu = s * (1.f / 128.f);
    float var = sq * (1.f / 128.f) - mu * mu;
    float inv = rsqrtf(var + LN_EPS);
    int row = m0 + r + 8 * hi;
#pragma unroll
    for (int j = 0; j < 8; ++j) {
      int n = j * 16 + lm;
      out[(size_t)row * H_ + n] = gamma[n] * (acc[j][r] - mu) * inv + beta[n];
    }
  }
}

extern "C" void kernel_launch(void* const* d_in, const int* in_sizes, int n_in,
                              void* d_out, int out_size, void* d_ws, size_t ws_size,
                              hipStream_t stream) {
  const float* enc   = (const float*)d_in[0];
  const int*   mask  = (const int*)d_in[1];
  const float* Wq    = (const float*)d_in[2];
  const float* Wk    = (const float*)d_in[3];
  const float* Wv    = (const float*)d_in[4];
  const float* Wo    = (const float*)d_in[5];
  const float* gamma = (const float*)d_in[6];
  const float* beta  = (const float*)d_in[7];
  float* out = (float*)d_out;

  // workspace layout (bf16 elements), total ~143 MB
  __bf16* ws   = (__bf16*)d_ws;
  __bf16* encB = ws;                                   // BS*H      = 4,194,304
  __bf16* WqT  = encB + (size_t)BS_ * H_;              // 65,536
  __bf16* WkT  = WqT + (size_t)H_ * NHH_;
  __bf16* WvT  = WkT + (size_t)H_ * NHH_;
  __bf16* WoT  = WvT + (size_t)H_ * NHH_;
  __bf16* Qb   = WoT + (size_t)H_ * NHH_;              // B*NH*S*H  = 16,777,216
  __bf16* Kb   = Qb + (size_t)B_ * NH_ * S_ * H_;
  __bf16* Vt   = Kb + (size_t)B_ * NH_ * S_ * H_;
  __bf16* Ctxb = Vt + (size_t)B_ * NH_ * S_ * H_;      // BS*NHH   = 16,777,216

  k_cast<<<(BS_ * H_ + 255) / 256, 256, 0, stream>>>(enc, encB, BS_ * H_);
  k_tcast<<<(H_ * NHH_ + 255) / 256, 256, 0, stream>>>(Wq, WqT, H_, NHH_);
  k_tcast<<<(H_ * NHH_ + 255) / 256, 256, 0, stream>>>(Wk, WkT, H_, NHH_);
  k_tcast<<<(H_ * NHH_ + 255) / 256, 256, 0, stream>>>(Wv, WvT, H_, NHH_);
  k_tcast<<<(NHH_ * H_ + 255) / 256, 256, 0, stream>>>(Wo, WoT, NHH_, H_);

  dim3 gq(BS_ / 64, NHH_ / 64, 3);
  k_qkv<<<gq, 128, 0, stream>>>(encB, WqT, WkT, WvT, Qb, Kb, Vt);

  dim3 ga(S_ / 64, NH_, B_);
  k_attn<<<ga, 128, 0, stream>>>(Qb, Kb, Vt, mask, Ctxb);

  k_out<<<BS_ / 64, 128, 0, stream>>>(Ctxb, WoT, enc, gamma, beta, out);
}
